// BinaryDilGroupConv_83803401879817
// MI455X (gfx1250) — 8-bit matrix operand order repaired and checked by reading, compiled, not run on hardware
//
#include <hip/hip_runtime.h>
#include <hip/hip_bf16.h>
#include <stdint.h>

#define B_    32
#define C_    256
#define H_    56
#define W_    56
#define HP_   58
#define HO_   28
#define WO_   28
#define G_    4
#define CPG_  64
#define NPIX  (B_*HO_*WO_)
#define NOUT  (B_*C_*HO_*WO_)
#define NIN   (B_*C_*H_*W_)
#define NPAD  (B_*G_*HP_*HP_*CPG_)

typedef __attribute__((ext_vector_type(8))) int          v8i;
typedef __attribute__((ext_vector_type(4))) unsigned int u32x4;

__device__ __forceinline__ signed char sgn8(float v) {
  return v > 0.f ? (signed char)1 : (v < 0.f ? (signed char)-1 : (signed char)0);
}


__global__ void zero_stats_kernel(float* st) {
  if (threadIdx.x < 1024) st[threadIdx.x] = 0.f;
}

__global__ void binx_kernel(const float* __restrict__ x, signed char* __restrict__ xb) {
  int i = blockIdx.x * 256 + threadIdx.x;
  if (i >= NPAD) return;
  int cg = i & 63; int t = i >> 6;
  int wp = t % HP_; t /= HP_;
  int hp = t % HP_; t /= HP_;
  int g = t % G_;  int b = t / G_;
  signed char v = 0;
  if (hp >= 1 && hp <= H_ && wp >= 1 && wp <= W_) {
    int c = g * 64 + cg;
    v = sgn8(x[((size_t)(b * C_ + c) * H_ + (hp - 1)) * W_ + (wp - 1)]);
  }
  xb[i] = v;
}

__global__ void maxpool_kernel(const float* __restrict__ x, float* __restrict__ sc) {
  int i = blockIdx.x * 256 + threadIdx.x;
  if (i >= NOUT) return;
  int wo = i % WO_; int t = i / WO_;
  int ho = t % HO_; t /= HO_;
  int c = t % C_;  int b = t / C_;
  const float* xp = x + ((size_t)b * C_ + c) * (H_ * W_);
  float m = -__builtin_inff();
  #pragma unroll
  for (int dy = 0; dy < 3; ++dy) {
    int ih = 2 * ho - 1 + dy;
    if (ih < 0 || ih >= H_) continue;
    #pragma unroll
    for (int dx = 0; dx < 3; ++dx) {
      int iw = 2 * wo - 1 + dx;
      if (iw < 0 || iw >= W_) continue;
      m = fmaxf(m, xp[ih * W_ + iw]);
    }
  }
  sc[i] = m;
}

__global__ void binw1_kernel(const float* __restrict__ w1, signed char* __restrict__ wA1) {
  int i = blockIdx.x * 256 + threadIdx.x;
  if (i >= C_ * CPG_ * 9) return;
  int tap = i % 9; int t = i / 9;
  int cin = t % CPG_; int co = t / CPG_;
  wA1[(size_t)co * 576 + tap * 64 + cin] = sgn8(w1[i]);
}

__global__ void binw2_kernel(const float* __restrict__ w2, signed char* __restrict__ wA2) {
  int i = blockIdx.x * 256 + threadIdx.x;
  if (i >= C_ * C_) return;
  wA2[i] = sgn8(w2[i]);
}


__device__ __forceinline__ void async_copy_to_lds(signed char* lds, const signed char* gsrc,
                                                  int tid, int nthreads, int total_qwords) {
  for (int idx = tid; idx < total_qwords; idx += nthreads) {
    int off = idx * 8;
    unsigned laddr = (unsigned)(unsigned long long)(void*)(lds + off);
    unsigned long long ga = (unsigned long long)(gsrc + off);
    asm volatile("global_load_async_to_lds_b64 %0, %1, off" :: "v"(laddr), "v"(ga) : "memory");
  }
  asm volatile("s_wait_asynccnt 0" ::: "memory");
}

union AF { v8i v; unsigned long long d[4]; };
union BF { v8i v; u32x4 q[2]; };

__device__ __forceinline__ void load_afrag(AF& af, const signed char* ar) {
  af.d[0] = *(const unsigned long long*)(ar);
  af.d[1] = *(const unsigned long long*)(ar + 8);
  af.d[2] = *(const unsigned long long*)(ar + 32);
  af.d[3] = *(const unsigned long long*)(ar + 40);
}

__global__ __launch_bounds__(256) void conv1_wmma_kernel(
    const signed char* __restrict__ xb, const signed char* __restrict__ wA1,
    float* __restrict__ raw1, float* __restrict__ st)
{
  __shared__ __align__(16) signed char shA[16 * 576];
  const int tid  = threadIdx.x;
  const int wave = tid >> 5, lane = tid & 31;
  const int half = lane >> 4, l15 = lane & 15;
  const int g = blockIdx.z, mt = blockIdx.y;

  async_copy_to_lds(shA, wA1 + (size_t)(g * 64 + mt * 16) * 576, tid, 256, 1152);
  __syncthreads();

  const int n0 = blockIdx.x * 256 + wave * 32 + l15;
  const int n1 = n0 + 16;
  const int b0 = n0 / (HO_ * WO_), r0 = n0 % (HO_ * WO_);
  const int b1 = n1 / (HO_ * WO_), r1 = n1 % (HO_ * WO_);
  const int ho0 = r0 / WO_, wo0 = r0 % WO_;
  const int ho1 = r1 / WO_, wo1 = r1 % WO_;

  const signed char* p0 =
      xb + (((size_t)(b0 * G_ + g) * (HP_ * HP_) + (2 * ho0) * HP_ + 2 * wo0) << 6) + half * 16;
  const signed char* p1 =
      xb + (((size_t)(b1 * G_ + g) * (HP_ * HP_) + (2 * ho1) * HP_ + 2 * wo1) << 6) + half * 16;

  v8i acc0 = {0, 0, 0, 0, 0, 0, 0, 0};
  v8i acc1 = {0, 0, 0, 0, 0, 0, 0, 0};

  #pragma unroll
  for (int t = 0; t < 9; ++t) {
    const int toff = ((t / 3) * HP_ + (t % 3)) * 64;
    AF af; load_afrag(af, shA + l15 * 576 + t * 64 + half * 16);
    BF bf0, bf1;
    bf0.q[0] = *(const u32x4*)(p0 + toff);
    bf0.q[1] = *(const u32x4*)(p0 + toff + 32);
    bf1.q[0] = *(const u32x4*)(p1 + toff);
    bf1.q[1] = *(const u32x4*)(p1 + toff + 32);
    acc0 = __builtin_amdgcn_wmma_i32_16x16x64_iu8(true, af.v, true, bf0.v, acc0, false, false);
    acc1 = __builtin_amdgcn_wmma_i32_16x16x64_iu8(true, af.v, true, bf1.v, acc1, false, false);
  }

  const int cbase = g * 64 + mt * 16 + half * 8;
  float v0s[8], v1s[8];
  #pragma unroll
  for (int r = 0; r < 8; ++r) {
    float v0 = (float)acc0[r], v1 = (float)acc1[r];
    v0s[r] = v0; v1s[r] = v1;
    raw1[((size_t)b0 * C_ + (cbase + r)) * (HO_ * WO_) + ho0 * WO_ + wo0] = v0;
    raw1[((size_t)b1 * C_ + (cbase + r)) * (HO_ * WO_) + ho1 * WO_ + wo1] = v1;
  }
  float* sum1 = st; float* sqs1 = st + C_;
  #pragma unroll
  for (int r = 0; r < 8; ++r) {
    float s = v0s[r] + v1s[r];
    float q = v0s[r] * v0s[r] + v1s[r] * v1s[r];
    #pragma unroll
    for (int o = 1; o < 16; o <<= 1) { s += __shfl_xor(s, o, 32); q += __shfl_xor(q, o, 32); }
    if (l15 == 0) { atomicAdd(&sum1[cbase + r], s); atomicAdd(&sqs1[cbase + r], q); }
  }
}

__global__ void bn1_kernel(const float* __restrict__ raw1, const float* __restrict__ sc,
                           const float* __restrict__ st, const float* __restrict__ g1,
                           const float* __restrict__ b1, float* __restrict__ x1,
                           signed char* __restrict__ x1b)
{
  int i = blockIdx.x * 256 + threadIdx.x;
  if (i >= NOUT) return;
  int wo = i % WO_; int t = i / WO_;
  int ho = t % HO_; t /= HO_;
  int c = t % C_;  int b = t / C_;
  const float cnt = (float)NPIX;
  float mean = st[c] / cnt;
  float var  = st[C_ + c] / cnt - mean * mean;
  float inv  = rsqrtf(var + 1e-5f);
  float v = (raw1[i] - mean) * inv * g1[c] + b1[c] + sc[i];
  x1[i] = v;
  int n = b * (HO_ * WO_) + ho * WO_ + wo;
  x1b[(size_t)n * C_ + c] = sgn8(v);
}

__global__ __launch_bounds__(256) void conv2_wmma_kernel(
    const signed char* __restrict__ x1b, const signed char* __restrict__ wA2,
    float* __restrict__ raw2, float* __restrict__ st)
{
  __shared__ __align__(16) signed char shA[16 * 256];
  const int tid  = threadIdx.x;
  const int wave = tid >> 5, lane = tid & 31;
  const int half = lane >> 4, l15 = lane & 15;
  const int mt = blockIdx.y;

  async_copy_to_lds(shA, wA2 + (size_t)mt * 16 * 256, tid, 256, 512);
  __syncthreads();

  const int n0 = blockIdx.x * 256 + wave * 32 + l15;
  const int n1 = n0 + 16;
  const int b0 = n0 / (HO_ * WO_), r0 = n0 % (HO_ * WO_);
  const int b1 = n1 / (HO_ * WO_), r1 = n1 % (HO_ * WO_);
  const int ho0 = r0 / WO_, wo0 = r0 % WO_;
  const int ho1 = r1 / WO_, wo1 = r1 % WO_;

  const signed char* p0 = x1b + (size_t)n0 * 256 + half * 16;
  const signed char* p1 = x1b + (size_t)n1 * 256 + half * 16;

  v8i acc0 = {0, 0, 0, 0, 0, 0, 0, 0};
  v8i acc1 = {0, 0, 0, 0, 0, 0, 0, 0};
  #pragma unroll
  for (int s = 0; s < 4; ++s) {
    AF af; load_afrag(af, shA + l15 * 256 + s * 64 + half * 16);
    BF bf0, bf1;
    bf0.q[0] = *(const u32x4*)(p0 + s * 64);
    bf0.q[1] = *(const u32x4*)(p0 + s * 64 + 32);
    bf1.q[0] = *(const u32x4*)(p1 + s * 64);
    bf1.q[1] = *(const u32x4*)(p1 + s * 64 + 32);
    acc0 = __builtin_amdgcn_wmma_i32_16x16x64_iu8(true, af.v, true, bf0.v, acc0, false, false);
    acc1 = __builtin_amdgcn_wmma_i32_16x16x64_iu8(true, af.v, true, bf1.v, acc1, false, false);
  }

  const int cbase = mt * 16 + half * 8;
  float v0s[8], v1s[8];
  #pragma unroll
  for (int r = 0; r < 8; ++r) {
    float v0 = (float)acc0[r], v1 = (float)acc1[r];
    v0s[r] = v0; v1s[r] = v1;
    raw2[((size_t)b0 * C_ + (cbase + r)) * (HO_ * WO_) + ho0 * WO_ + wo0] = v0;
    raw2[((size_t)b1 * C_ + (cbase + r)) * (HO_ * WO_) + ho1 * WO_ + wo1] = v1;
  }
  float* sum2 = st; float* sqs2 = st + C_;
  #pragma unroll
  for (int r = 0; r < 8; ++r) {
    float s = v0s[r] + v1s[r];
    float q = v0s[r] * v0s[r] + v1s[r] * v1s[r];
    #pragma unroll
    for (int o = 1; o < 16; o <<= 1) { s += __shfl_xor(s, o, 32); q += __shfl_xor(q, o, 32); }
    if (l15 == 0) { atomicAdd(&sum2[cbase + r], s); atomicAdd(&sqs2[cbase + r], q); }
  }
}

__global__ void bnfinal_kernel(const float* __restrict__ raw2, const float* __restrict__ x1,
                               const float* __restrict__ st, const float* __restrict__ g2,
                               const float* __restrict__ b2, float* __restrict__ out)
{
  int i = blockIdx.x * 256 + threadIdx.x;
  if (i >= NOUT) return;
  int t = i / (HO_ * WO_);
  int c = t % C_;
  const float cnt = (float)NPIX;
  float mean = st[c] / cnt;
  float var  = st[C_ + c] / cnt - mean * mean;
  float inv  = rsqrtf(var + 1e-5f);
  out[i] = (raw2[i] - mean) * inv * g2[c] + b2[c] + x1[i];
}


extern "C" void kernel_launch(void* const* d_in, const int* in_sizes, int n_in,
                              void* d_out, int out_size, void* d_ws, size_t ws_size,
                              hipStream_t stream) {
  const float* x  = (const float*)d_in[0];
  const float* w1 = (const float*)d_in[1];
  const float* g1 = (const float*)d_in[2];
  const float* b1 = (const float*)d_in[3];
  const float* w2 = (const float*)d_in[4];
  const float* g2 = (const float*)d_in[5];
  const float* b2 = (const float*)d_in[6];
  (void)in_sizes; (void)n_in; (void)out_size; (void)ws_size;

  char* ws = (char*)d_ws;
  const size_t SZ   = (size_t)NIN;
  const size_t XBSZ = (size_t)NPAD;
  signed char* xb  = (signed char*)(ws);
  float*       sc  = (float*)(ws + XBSZ);
  float*       raw1= (float*)(ws + XBSZ + SZ);
  float*       x1  = (float*)(ws + XBSZ + 2 * SZ);
  signed char* x1b = (signed char*)(ws + XBSZ + 3 * SZ);
  signed char* wA1 = (signed char*)(ws + XBSZ + 3 * SZ + (size_t)NOUT);
  signed char* wA2 = wA1 + (size_t)C_ * 576;
  float*       st  = (float*)((char*)wA2 + (size_t)C_ * C_);
  float*       raw2= (float*)xb;

  float* out = (float*)d_out;

  zero_stats_kernel<<<1, 1024, 0, stream>>>(st);
  binx_kernel<<<(NPAD + 255) / 256, 256, 0, stream>>>(x, xb);
  maxpool_kernel<<<(NOUT + 255) / 256, 256, 0, stream>>>(x, sc);
  binw1_kernel<<<(C_ * CPG_ * 9 + 255) / 256, 256, 0, stream>>>(w1, wA1);
  binw2_kernel<<<(C_ * C_ + 255) / 256, 256, 0, stream>>>(w2, wA2);

  conv1_wmma_kernel<<<dim3(NPIX / 256, 4, 4), 256, 0, stream>>>(xb, wA1, raw1, st);
  bn1_kernel<<<(NOUT + 255) / 256, 256, 0, stream>>>(raw1, sc, st, g1, b1, x1, x1b);
  conv2_wmma_kernel<<<dim3(NPIX / 256, 16, 1), 256, 0, stream>>>(x1b, wA2, raw2, st + 512);
  bnfinal_kernel<<<(NOUT + 255) / 256, 256, 0, stream>>>(raw2, x1, st + 512, g2, b2, out);
}
